// MutilSpatial_38809324487309
// MI455X (gfx1250) — matrix operand order repaired and checked by reading, not compiled, not run on hardware
//
#include <hip/hip_runtime.h>

typedef __attribute__((ext_vector_type(16))) _Float16 v16h;
typedef __attribute__((ext_vector_type(8)))  _Float16 v8h;
typedef __attribute__((ext_vector_type(2)))  _Float16 h2;
typedef __attribute__((ext_vector_type(8)))  float    v8f;

#define HW   6400   // 80*80
#define C1   256
#define C2   256
#define GT   64     // channels per group
#define NPOS 102400 // B*H*W = 16*6400

union V16 { v16h v; v8h h[2]; };

// ---------------------------------------------------------------------------
// Kernel 1: per-group channel squeeze  sq[bg][{mean,max}][hw]
// ---------------------------------------------------------------------------
__global__ __launch_bounds__(256) void squeeze_kernel(
    const float* __restrict__ x, float* __restrict__ sq)
{
    int idx = blockIdx.x * 256 + threadIdx.x;      // 0..409599  (bg*6400+hw)
    int hw  = idx % HW;
    int bg  = idx / HW;                            // 0..63
    const float* p = x + (size_t)bg * GT * HW + hw;
    float s = 0.f, mx = -3.402823466e38f;
    #pragma unroll 16
    for (int c = 0; c < GT; ++c) {
        float v = p[(size_t)c * HW];
        s += v;
        mx = fmaxf(mx, v);
    }
    sq[(size_t)bg * 2 * HW + hw]       = s * (1.0f / 64.0f);
    sq[(size_t)bg * 2 * HW + HW + hw]  = mx;
}

// ---------------------------------------------------------------------------
// Kernel 2: attention a = sigmoid(lrelu(bn(conv3))) + sigmoid(lrelu(bn(conv7)))
// ---------------------------------------------------------------------------
__global__ __launch_bounds__(256) void attn_kernel(
    const float* __restrict__ sq,
    const float* __restrict__ w3,
    const float* __restrict__ g1, const float* __restrict__ b1,
    const float* __restrict__ m1, const float* __restrict__ v1,
    const float* __restrict__ w7,
    const float* __restrict__ g2, const float* __restrict__ b2,
    const float* __restrict__ m2, const float* __restrict__ v2,
    float* __restrict__ aout)
{
    int idx = blockIdx.x * 256 + threadIdx.x;      // bg*6400 + hw
    int hw  = idx % HW;
    int bg  = idx / HW;
    int yy  = hw / 80, xx = hw % 80;
    const float* s0 = sq + (size_t)bg * 2 * HW;    // mean plane
    const float* s1 = s0 + HW;                     // max plane

    float c3 = 0.f;
    #pragma unroll
    for (int dy = -1; dy <= 1; ++dy) {
        int y = yy + dy;
        if (y < 0 || y >= 80) continue;
        #pragma unroll
        for (int dx = -1; dx <= 1; ++dx) {
            int x2 = xx + dx;
            if (x2 < 0 || x2 >= 80) continue;
            int o = y * 80 + x2;
            int wi = (dy + 1) * 3 + (dx + 1);
            c3 += w3[wi] * s0[o] + w3[9 + wi] * s1[o];
        }
    }
    float c7 = 0.f;
    #pragma unroll
    for (int dy = -3; dy <= 3; ++dy) {
        int y = yy + dy;
        if (y < 0 || y >= 80) continue;
        #pragma unroll
        for (int dx = -3; dx <= 3; ++dx) {
            int x2 = xx + dx;
            if (x2 < 0 || x2 >= 80) continue;
            int o = y * 80 + x2;
            int wi = (dy + 3) * 7 + (dx + 3);
            c7 += w7[wi] * s0[o] + w7[49 + wi] * s1[o];
        }
    }
    float sc1 = g1[0] * rsqrtf(v1[0] + 1e-5f);
    float y1  = (c3 - m1[0]) * sc1 + b1[0];
    y1 = y1 > 0.f ? y1 : 0.1f * y1;
    float a1 = 1.0f / (1.0f + __expf(-y1));

    float sc2 = g2[0] * rsqrtf(v2[0] + 1e-5f);
    float y2  = (c7 - m2[0]) * sc2 + b2[0];
    y2 = y2 > 0.f ? y2 : 0.1f * y2;
    float a2 = 1.0f / (1.0f + __expf(-y2));

    aout[idx] = a1 + a2;
}

// ---------------------------------------------------------------------------
// Kernel 3: convert final 1x1 conv weight [C2][C1] to f16
// ---------------------------------------------------------------------------
__global__ __launch_bounds__(256) void wcvt_kernel(
    const float* __restrict__ w, _Float16* __restrict__ wf)
{
    int i = blockIdx.x * 256 + threadIdx.x;        // 65536 elems
    wf[i] = (_Float16)w[i];
}

// ---------------------------------------------------------------------------
// Kernel 4: fused feat = x*a  +  1x1 conv GEMM via WMMA f16 -> f32
//   A = weights (M=16 out-ch x K=32 f16), B = feat (K=32 x N=16 positions)
//   Block: 256 thr (8 waves); tile = 64 positions x 256 out-channels, K=256.
//   K-loop kept at partial unroll to stay <256 VGPRs (no s_set_vgpr_msb).
// ---------------------------------------------------------------------------
__global__ __launch_bounds__(256) void gemm_kernel(
    const float*   __restrict__ x,
    const float*   __restrict__ a,
    const _Float16* __restrict__ wf,
    const float*   __restrict__ bias,
    float*         __restrict__ out)
{
    __shared__ __align__(16) _Float16 smemh[16384];   // 32KB: 8 ksteps x 4 ptiles x frag

    const int tid     = threadIdx.x;
    const int posBase = blockIdx.x * 64;           // flat b*6400+hw, 6400%64==0
    const int b       = posBase / HW;
    const int hwBase  = posBase % HW;

    // ---- stage feat tile (64 pos x 256 K) into LDS in B-fragment layout ----
    {
        const int posl = tid & 63;                 // position in tile
        const int kof  = tid >> 6;                 // 0..3
        const float* xb = x + (size_t)b * C1 * HW + hwBase + posl;
        const float* ab = a + (size_t)b * 4 * HW + hwBase + posl;
        const int t        = posl >> 4;            // 16-pos sub-tile
        const int lanebase = posl & 15;
        #pragma unroll 4
        for (int i = 0; i < 32; ++i) {
            int k  = (i * 4 + kof) * 2;            // even k, pair (k,k+1)
            float av = ab[(size_t)(k >> 6) * HW];  // a[b*4 + g][hw]
            float f0 = xb[(size_t)k * HW] * av;
            float f1 = xb[(size_t)(k + 1) * HW] * av;
            int s    = k >> 5;                     // k-step
            int kk   = k & 31;
            int lane = lanebase + (kk & 16);       // lanes16-31 hold K=16..31
            int j    = kk & 15;
            h2 pr; pr.x = (_Float16)f0; pr.y = (_Float16)f1;
            *(h2*)(smemh + s * 2048 + t * 512 + lane * 16 + j) = pr;
        }
    }
    __syncthreads();

    // ---- per-wave WMMA main loop ----
    const int wave  = tid >> 5;
    const int lane  = tid & 31;
    const int ml    = lane & 15;                   // A-matrix row (out-channel)
    const int ak0   = (lane >> 4) * 16;            // A: lanes16-31 hold K=16..31

    v8f C[2][4];
    #pragma unroll
    for (int m = 0; m < 2; ++m)
        #pragma unroll
        for (int t = 0; t < 4; ++t)
            C[m][t] = (v8f){0.f,0.f,0.f,0.f,0.f,0.f,0.f,0.f};

    const _Float16* wrow0 = wf + (size_t)(wave * 32 + ml) * C1;
    const _Float16* wrow1 = wrow0 + 16 * C1;

    #pragma unroll 2
    for (int s = 0; s < 8; ++s) {
        V16 A0, A1;
        A0.h[0] = *(const v8h*)(wrow0 + s * 32 + ak0);
        A0.h[1] = *(const v8h*)(wrow0 + s * 32 + ak0 + 8);
        A1.h[0] = *(const v8h*)(wrow1 + s * 32 + ak0);
        A1.h[1] = *(const v8h*)(wrow1 + s * 32 + ak0 + 8);
        #pragma unroll
        for (int t = 0; t < 4; ++t) {
            V16 Bf;
            const _Float16* bp = smemh + s * 2048 + t * 512 + lane * 16;
            Bf.h[0] = *(const v8h*)(bp);
            Bf.h[1] = *(const v8h*)(bp + 8);
            C[0][t] = __builtin_amdgcn_wmma_f32_16x16x32_f16(
                false, A0.v, false, Bf.v, (short)0, C[0][t], false, false);
            C[1][t] = __builtin_amdgcn_wmma_f32_16x16x32_f16(
                false, A1.v, false, Bf.v, (short)0, C[1][t], false, false);
        }
    }

    // ---- epilogue: bias + coalesced f32 stores ----
    const int nlane = lane & 15;                   // position within 16-tile
    const int mhalf = (lane >> 4) * 8;             // lanes16-31 -> M += 8
    float* outb = out + (size_t)b * C2 * HW + hwBase;

    float bv[2][8];
    #pragma unroll
    for (int m = 0; m < 2; ++m)
        #pragma unroll
        for (int j = 0; j < 8; ++j)
            bv[m][j] = bias[wave * 32 + m * 16 + mhalf + j];

    #pragma unroll
    for (int m = 0; m < 2; ++m) {
        int cbase = wave * 32 + m * 16 + mhalf;
        #pragma unroll
        for (int t = 0; t < 4; ++t) {
            #pragma unroll
            for (int j = 0; j < 8; ++j) {
                outb[(size_t)(cbase + j) * HW + t * 16 + nlane] = C[m][t][j] + bv[m][j];
            }
        }
    }
}

// ---------------------------------------------------------------------------
extern "C" void kernel_launch(void* const* d_in, const int* in_sizes, int n_in,
                              void* d_out, int out_size, void* d_ws, size_t ws_size,
                              hipStream_t stream)
{
    const float* x      = (const float*)d_in[0];
    const float* w3     = (const float*)d_in[1];
    const float* bn1g   = (const float*)d_in[2];
    const float* bn1b   = (const float*)d_in[3];
    const float* bn1m   = (const float*)d_in[4];
    const float* bn1v   = (const float*)d_in[5];
    const float* w7     = (const float*)d_in[6];
    const float* bn2g   = (const float*)d_in[7];
    const float* bn2b   = (const float*)d_in[8];
    const float* bn2m   = (const float*)d_in[9];
    const float* bn2v   = (const float*)d_in[10];
    const float* wfin   = (const float*)d_in[11];
    const float* bfin   = (const float*)d_in[12];
    float* out = (float*)d_out;

    char* ws = (char*)d_ws;
    float*    sq  = (float*)ws;                       // 64*2*6400*4 = 3,276,800 B
    float*    aw  = (float*)(ws + 3276800);           // 64*6400*4   = 1,638,400 B
    _Float16* wf  = (_Float16*)(ws + 4915200);        // 65536*2     =   131,072 B

    squeeze_kernel<<<1600, 256, 0, stream>>>(x, sq);
    attn_kernel<<<1600, 256, 0, stream>>>(sq, w3, bn1g, bn1b, bn1m, bn1v,
                                          w7, bn2g, bn2b, bn2m, bn2v, aw);
    wcvt_kernel<<<256, 256, 0, stream>>>(wfin, wf);
    gemm_kernel<<<1600, 256, 0, stream>>>(x, aw, wf, bfin, out);
}
